// GraphVAE_87333864997317
// MI455X (gfx1250) — matrix operand order repaired and checked by reading, not compiled, not run on hardware
//
#include <hip/hip_runtime.h>

typedef __attribute__((ext_vector_type(16))) __bf16 v16bf;
typedef __attribute__((ext_vector_type(8)))  float  v8f;

// ---------------- helpers ----------------
__device__ __forceinline__ unsigned short f2bf(float f) {
    unsigned int u = __float_as_uint(f);
    unsigned int r = u + 0x7FFFu + ((u >> 16) & 1u);   // round-to-nearest-even
    return (unsigned short)(r >> 16);
}

// ---------------- conversion kernels ----------------
__global__ void k_cvt_bf16(const float* __restrict__ x, unsigned short* __restrict__ y,
                           long long n) {
    long long i = blockIdx.x * (long long)blockDim.x + threadIdx.x;
    if (i < n) y[i] = f2bf(x[i]);
}

// Wt[fout, fin] = bf16(W[fin, fout])  (transpose so B-fragment loads are contiguous in K)
__global__ void k_wt_cvt(const float* __restrict__ w, unsigned short* __restrict__ wt,
                         int fin, int fout) {
    int i = blockIdx.x * blockDim.x + threadIdx.x;
    if (i >= fin * fout) return;
    int r = i / fout, c = i - r * fout;
    wt[(size_t)c * fin + r] = f2bf(w[i]);
}

// ---------------- degree / normalization ----------------
__global__ void k_zero(float* p, int n) {
    int i = blockIdx.x * blockDim.x + threadIdx.x;
    if (i < n) p[i] = 0.0f;
}
__global__ void k_deg(const int* __restrict__ dst, float* deg, int e) {
    int i = blockIdx.x * blockDim.x + threadIdx.x;
    if (i < e) atomicAdd(&deg[dst[i]], 1.0f);
}
__global__ void k_dinv(float* d, int n) {
    int i = blockIdx.x * blockDim.x + threadIdx.x;
    if (i < n) d[i] = rsqrtf(d[i] + 1.0f);   // +1 = self loop; always >= 1
}

// ---------------- WMMA GEMM: H[M,Nout] = A[M,K] * Bt[Nout,K]^T ----------------
// One wave computes a 16x64 output block: 4 accumulators share one A fragment,
// raising arithmetic intensity (10 b128 loads per 4 WMMAs instead of 16).
// Fused epilogue: optionally also writes Y = H * dinv[row]^2 (self-loop term of
// the GCN aggregation, which doubles as Y's zero-init).
__global__ void k_gemm_wmma(const unsigned short* __restrict__ A,   // bf16 [M,K]
                            const unsigned short* __restrict__ Bt,  // bf16 [Nout,K]
                            float* __restrict__ H,                  // f32 [M,Nout]
                            float* __restrict__ Y,                  // f32 [M,Nout] or nullptr
                            const float* __restrict__ dinv,
                            int M, int K, int Nout) {
    int wave = (int)((blockIdx.x * (unsigned)blockDim.x + threadIdx.x) >> 5);
    int lane = threadIdx.x & 31;
    int nblk = Nout >> 6;                     // 64-wide blocks (Nout in {64,128,256})
    int total = (M >> 4) * nblk;
    if (wave >= total) return;                // wave-uniform: EXEC all-1s for WMMA
    int tm = wave / nblk;
    int tb = wave - tm * nblk;
    int m0 = tm << 4, n0 = tb << 6;
    int half = lane >> 4, l16 = lane & 15;

    union U { v16bf v; uint4 u[2]; };
    U a, b0, b1, b2, b3;
    v8f acc0 = {0.f,0.f,0.f,0.f,0.f,0.f,0.f,0.f};
    v8f acc1 = acc0, acc2 = acc0, acc3 = acc0;

    const unsigned short* arow = A  + (size_t)(m0 + l16) * K;
    const unsigned short* br0  = Bt + (size_t)(n0 +  0 + l16) * K;
    const unsigned short* br1  = Bt + (size_t)(n0 + 16 + l16) * K;
    const unsigned short* br2  = Bt + (size_t)(n0 + 32 + l16) * K;
    const unsigned short* br3  = Bt + (size_t)(n0 + 48 + l16) * K;
    const int ak = half * 16;                 // A: K in {ak..ak+15} contiguous
    const int bk = half * 16;                 // B: K in {bk..bk+15} contiguous

    for (int k0 = 0; k0 < K; k0 += 32) {
        a.u[0]  = *(const uint4*)(arow + k0 + ak);
        a.u[1]  = *(const uint4*)(arow + k0 + ak + 8);
        b0.u[0] = *(const uint4*)(br0 + k0 + bk);
        b0.u[1] = *(const uint4*)(br0 + k0 + bk + 8);
        b1.u[0] = *(const uint4*)(br1 + k0 + bk);
        b1.u[1] = *(const uint4*)(br1 + k0 + bk + 8);
        b2.u[0] = *(const uint4*)(br2 + k0 + bk);
        b2.u[1] = *(const uint4*)(br2 + k0 + bk + 8);
        b3.u[0] = *(const uint4*)(br3 + k0 + bk);
        b3.u[1] = *(const uint4*)(br3 + k0 + bk + 8);
        acc0 = __builtin_amdgcn_wmma_f32_16x16x32_bf16(false, a.v, false, b0.v, (short)0, acc0, false, false);
        acc1 = __builtin_amdgcn_wmma_f32_16x16x32_bf16(false, a.v, false, b1.v, (short)0, acc1, false, false);
        acc2 = __builtin_amdgcn_wmma_f32_16x16x32_bf16(false, a.v, false, b2.v, (short)0, acc2, false, false);
        acc3 = __builtin_amdgcn_wmma_f32_16x16x32_bf16(false, a.v, false, b3.v, (short)0, acc3, false, false);
    }

    // D layout: VGPR i -> (M = m0 + i + 8*half, N = n0 + noff + l16)
    float dv2[8];
#pragma unroll
    for (int i = 0; i < 8; ++i) dv2[i] = 0.0f;
    if (Y) {
#pragma unroll
        for (int i = 0; i < 8; ++i) {
            float dv = dinv[m0 + half * 8 + i];
            dv2[i] = dv * dv;
        }
    }
    size_t rowbase = (size_t)(m0 + half * 8) * Nout + n0 + l16;
    auto store_tile = [&](const v8f& acc, int noff) {
        float* h = H + rowbase + noff;
#pragma unroll
        for (int i = 0; i < 8; ++i) h[(size_t)i * Nout] = acc[i];
        if (Y) {
            float* y = Y + rowbase + noff;
#pragma unroll
            for (int i = 0; i < 8; ++i) y[(size_t)i * Nout] = acc[i] * dv2[i];
        }
    };
    store_tile(acc0, 0);
    store_tile(acc1, 16);
    store_tile(acc2, 32);
    store_tile(acc3, 48);
}

// ---------------- aggregation ----------------
// per-edge scatter: Y[dst] += H[src] * dinv[src]*dinv[dst], 4 features per thread
__global__ void k_edges(const float* __restrict__ H, const float* __restrict__ dinv,
                        const int* __restrict__ src, const int* __restrict__ dst,
                        float* __restrict__ Y, int E, int F, int cshift) {
    long long idx = blockIdx.x * (long long)blockDim.x + threadIdx.x;
    long long total = (long long)E << cshift;          // E * (F/4)
    if (idx >= total) return;
    int e = (int)(idx >> cshift);
    int c = (int)(idx & ((1 << cshift) - 1));
    int s = src[e], d = dst[e];
    float nrm = dinv[s] * dinv[d];
    const float4 h = *(const float4*)(H + (size_t)s * F + c * 4);
    float* y = Y + (size_t)d * F + c * 4;
    atomicAdd(y + 0, h.x * nrm);
    atomicAdd(y + 1, h.y * nrm);
    atomicAdd(y + 2, h.z * nrm);
    atomicAdd(y + 3, h.w * nrm);
}

// ---------------- bias + activation epilogues ----------------
// Intermediate layers: relu(X + b) written directly as bf16 staging for the next GEMM.
__global__ void k_biasrelu_bf16(const float* __restrict__ X, const float* __restrict__ b,
                                unsigned short* __restrict__ Y, int F) {
    int node = blockIdx.x;
    size_t base = (size_t)node * F;
    for (int f = threadIdx.x; f < F; f += blockDim.x) {
        float v = fmaxf(X[base + f] + b[f], 0.0f);
        Y[base + f] = f2bf(v);
    }
}

// Final/f32 outputs. act: 0 = none, 2 = sigmoid
__global__ void k_biasact_f32(const float* __restrict__ X, const float* __restrict__ b,
                              float* __restrict__ Y, int F, int act) {
    int node = blockIdx.x;
    size_t base = (size_t)node * F;
    for (int f = threadIdx.x; f < F; f += blockDim.x) {
        float v = X[base + f] + b[f];
        if (act == 2) v = 1.0f / (1.0f + __expf(-v));
        Y[base + f] = v;
    }
}

// z = mu + log_var * eps, written directly as bf16 staging for the decoder GEMM
__global__ void k_z_bf16(const float* __restrict__ mu, const float* __restrict__ lv,
                         const float* __restrict__ eps, unsigned short* __restrict__ z,
                         long long n) {
    long long i = blockIdx.x * (long long)blockDim.x + threadIdx.x;
    if (i < n) z[i] = f2bf(fmaf(lv[i], eps[i], mu[i]));
}

// ---------------- host-side composition ----------------
static inline int log2F4(int F) { return (F == 256) ? 6 : (F == 128) ? 5 : 4; }

static void gemm_stage(int Fin, int Fout, const float* W,
                       const unsigned short* X16, unsigned short* Wt16,
                       float* Hbuf, float* Ybuf, const float* dinv,
                       int n, hipStream_t s) {
    int wcnt = Fin * Fout;
    k_wt_cvt<<<(wcnt + 255) / 256, 256, 0, s>>>(W, Wt16, Fin, Fout);
    int waves = (n / 16) * (Fout / 64);
    k_gemm_wmma<<<(waves + 7) / 8, 256, 0, s>>>(X16, Wt16, Hbuf, Ybuf, dinv, n, Fin, Fout);
}

// One GCN layer. Input activations already staged as bf16 in X16 [n, Fin].
// If f32out == nullptr: relu + write bf16 back into X16 (next layer's input).
// Else: apply `act` and write f32 into f32out.
static void gcn_layer(int Fin, int Fout, const float* W, const float* b,
                      float* f32out, int act,
                      unsigned short* X16, unsigned short* Wt16,
                      float* Hbuf, float* Ybuf,
                      const float* dinv, const int* src, const int* dst,
                      int n, int e, hipStream_t s) {
    gemm_stage(Fin, Fout, W, X16, Wt16, Hbuf, Ybuf, dinv, n, s);
    int cs = log2F4(Fout);
    long long tot = (long long)e << cs;
    k_edges<<<(unsigned)((tot + 255) / 256), 256, 0, s>>>(Hbuf, dinv, src, dst, Ybuf, e, Fout, cs);
    if (f32out)
        k_biasact_f32<<<n, 256, 0, s>>>(Ybuf, b, f32out, Fout, act);
    else
        k_biasrelu_bf16<<<n, 256, 0, s>>>(Ybuf, b, X16, Fout);
}

extern "C" void kernel_launch(void* const* d_in, const int* in_sizes, int n_in,
                              void* d_out, int out_size, void* d_ws, size_t ws_size,
                              hipStream_t stream) {
    const float* x   = (const float*)d_in[0];
    const int*   ei  = (const int*)  d_in[1];
    const float* eps = (const float*)d_in[2];
    const float* w1  = (const float*)d_in[3];
    const float* b1  = (const float*)d_in[4];
    const float* w2  = (const float*)d_in[5];
    const float* b2  = (const float*)d_in[6];
    const float* wmu = (const float*)d_in[7];
    const float* bmu = (const float*)d_in[8];
    const float* wlv = (const float*)d_in[9];
    const float* blv = (const float*)d_in[10];
    const float* w3  = (const float*)d_in[11];
    const float* b3  = (const float*)d_in[12];
    const float* w4  = (const float*)d_in[13];
    const float* b4  = (const float*)d_in[14];
    const float* w5  = (const float*)d_in[15];
    const float* b5  = (const float*)d_in[16];

    const int n = in_sizes[0] / 128;   // 100000
    const int e = in_sizes[1] / 2;     // 3200000
    const int* src = ei;
    const int* dst = ei + e;

    // workspace carve-up
    char* w = (char*)d_ws;
    size_t o = 0;
    float* dinv = (float*)(w + o);  o += ((size_t)n * 4 + 255) & ~(size_t)255;
    float* Hbuf = (float*)(w + o);  o += (size_t)n * 256 * 4;
    float* Ybuf = (float*)(w + o);  o += (size_t)n * 256 * 4;
    unsigned short* X16  = (unsigned short*)(w + o);  o += (size_t)n * 256 * 2;
    unsigned short* Wt16 = (unsigned short*)(w + o);  o += (size_t)256 * 256 * 2;

    // output carve-up: (recon[N,128], mu[N,64], log_var[N,64])
    float* recon  = (float*)d_out;
    float* mu_out = recon + (size_t)n * 128;
    float* lv_out = mu_out + (size_t)n * 64;

    // ---- degree -> dinv ----
    k_zero<<<(n + 255) / 256, 256, 0, stream>>>(dinv, n);
    k_deg <<<(e + 255) / 256, 256, 0, stream>>>(dst, dinv, e);
    k_dinv<<<(n + 255) / 256, 256, 0, stream>>>(dinv, n);

    // ---- stage input x as bf16 ----
    {
        long long nx = (long long)n * 128;
        k_cvt_bf16<<<(unsigned)((nx + 255) / 256), 256, 0, stream>>>(x, X16, nx);
    }

    // ---- encoder (bf16 chained through X16) ----
    gcn_layer(128, 256, w1, b1, nullptr, 1, X16, Wt16, Hbuf, Ybuf, dinv, src, dst, n, e, stream);
    gcn_layer(256, 128, w2, b2, nullptr, 1, X16, Wt16, Hbuf, Ybuf, dinv, src, dst, n, e, stream);

    // ---- mu / log_var (dense linears, no aggregation, no self-loop epilogue) ----
    gemm_stage(128, 64, wmu, X16, Wt16, Hbuf, nullptr, dinv, n, stream);
    k_biasact_f32<<<n, 256, 0, stream>>>(Hbuf, bmu, mu_out, 64, 0);
    gemm_stage(128, 64, wlv, X16, Wt16, Hbuf, nullptr, dinv, n, stream);
    k_biasact_f32<<<n, 256, 0, stream>>>(Hbuf, blv, lv_out, 64, 0);

    // ---- reparameterize: z = mu + log_var * eps -> bf16 staging ----
    {
        long long nz = (long long)n * 64;
        k_z_bf16<<<(unsigned)((nz + 255) / 256), 256, 0, stream>>>(mu_out, lv_out, eps, X16, nz);
    }

    // ---- decoder ----
    gcn_layer( 64, 128, w3, b3, nullptr, 1, X16, Wt16, Hbuf, Ybuf, dinv, src, dst, n, e, stream);
    gcn_layer(128, 256, w4, b4, nullptr, 1, X16, Wt16, Hbuf, Ybuf, dinv, src, dst, n, e, stream);
    gcn_layer(256, 128, w5, b5, recon,   2, X16, Wt16, Hbuf, Ybuf, dinv, src, dst, n, e, stream);
}
